// RuleParse_33663953666597
// MI455X (gfx1250) — matrix operand order repaired and checked by reading, not compiled, not run on hardware
//
#include <hip/hip_runtime.h>
#include <hip/hip_bf16.h>

typedef __attribute__((ext_vector_type(16))) __bf16 v16bf;
typedef __attribute__((ext_vector_type(8)))  float  v8f;

#define WAVE 32

// ---------- order-preserving float<->uint encoding for atomic max ----------
__device__ __forceinline__ unsigned enc_f32(float f) {
  unsigned u = __float_as_uint(f);
  return (u & 0x80000000u) ? ~u : (u | 0x80000000u);
}
__device__ __forceinline__ float dec_f32(unsigned u) {
  u = (u & 0x80000000u) ? (u & 0x7fffffffu) : ~u;
  return __uint_as_float(u);
}

// ---------- generic fill ----------
__global__ void fill_f32(float* p, float v, long long n) {
  long long i = (long long)blockIdx.x * blockDim.x + threadIdx.x;
  if (i < n) p[i] = v;
}

// =====================================================================================
// Fragment-major packing for WMMA bf16 GEMM.
// A fragment (16x32, per ISA): lane = 16*half + (m&15); element pair p (0..7) maps to
//   kb = (p<4 ? 2p : 16+2(p-4)) + 8*half, elements (2p,2p+1) = kb, kb+1.
// B fragment (32x16): lane = 16*half + (n&15); element i maps to k = (i<8 ? i : 16+(i-8)) + 8*half.
// Packed layout: frag[tileRow * tilesK + kt] is 512 contiguous bf16; lane l owns
// elements [l*16, l*16+16) -> each lane loads 32 contiguous bytes (2x b128).
// =====================================================================================

__global__ void pack_a_frag(const float* __restrict__ A, __bf16* __restrict__ Apk,
                            int M, int K, int tilesK, long long total)
{
  long long idx = (long long)blockIdx.x * blockDim.x + threadIdx.x;
  if (idx >= total) return;
  int frag = (int)(idx >> 9);
  int r = (int)(idx & 511);
  int lane = r >> 4, i = r & 15;
  int tm = frag / tilesK, kt = frag % tilesK;
  int half = lane >> 4;
  int m = tm * 16 + (lane & 15);
  int p = i >> 1, odd = i & 1;
  int kb = ((p < 4) ? 2 * p : 16 + 2 * (p - 4)) + half * 8 + odd;
  int k = kt * 32 + kb;
  float v = (m < M && k < K) ? A[(size_t)m * K + k] : 0.f;
  Apk[idx] = (__bf16)v;
}

__global__ void pack_b_frag(const float* __restrict__ B, __bf16* __restrict__ Bpk,
                            int K, int N, int tilesK, long long total)
{
  long long idx = (long long)blockIdx.x * blockDim.x + threadIdx.x;
  if (idx >= total) return;
  int frag = (int)(idx >> 9);
  int r = (int)(idx & 511);
  int lane = r >> 4, i = r & 15;
  int tn = frag / tilesK, kt = frag % tilesK;
  int half = lane >> 4;
  int n = tn * 16 + (lane & 15);
  int k = kt * 32 + ((i < 8) ? i : 16 + (i - 8)) + half * 8;
  float v = (k < K && n < N) ? B[(size_t)k * N + n] : 0.f;
  Bpk[idx] = (__bf16)v;
}

// ---------- WMMA GEMM on packed fragments: C = A@B + bias, optional ReLU ----------
__global__ void gemm_wmma_packed(const v16bf* __restrict__ Apk, const v16bf* __restrict__ Bpk,
                                 const float* __restrict__ bias, float* __restrict__ C,
                                 int M, int N, int tilesK, int act)
{
  int wave = threadIdx.x >> 5;
  int lane = threadIdx.x & 31;
  int tilesN = (N + 15) >> 4;
  int tilesM = (M + 15) >> 4;
  int t = blockIdx.x * (blockDim.x >> 5) + wave;
  if (t >= tilesM * tilesN) return;
  int tm = t / tilesN, tn = t % tilesN;
  // 32 v16bf per fragment; lane's slice is one v16bf (32 bytes, 2x b128 loads)
  const v16bf* Ap = Apk + (size_t)tm * tilesK * 32 + lane;
  const v16bf* Bp = Bpk + (size_t)tn * tilesK * 32 + lane;
  v8f acc = {0.f, 0.f, 0.f, 0.f, 0.f, 0.f, 0.f, 0.f};
  for (int kt = 0; kt < tilesK; ++kt) {
    v16bf a = Ap[(size_t)kt * 32];
    v16bf b = Bp[(size_t)kt * 32];
    acc = __builtin_amdgcn_wmma_f32_16x16x32_bf16(false, a, false, b, (short)0, acc,
                                                  false, false);
  }
  int half = lane >> 4;
  int ncol = tn * 16 + (lane & 15);
  float bb = (ncol < N) ? bias[ncol] : 0.f;
#pragma unroll
  for (int r = 0; r < 8; ++r) {
    int m = tm * 16 + r + half * 8;
    if (m < M && ncol < N) {
      float v = acc[r] + bb;
      if (act == 1) v = fmaxf(v, 0.f);
      C[(size_t)m * N + ncol] = v;
    }
  }
}

// ---------- self-loop edge_attr mean: count + sum per dst ----------
__global__ void edge_count_sum(const int* __restrict__ dst, const float* __restrict__ ea,
                               float* ea_sum, float* cnt, int E)
{
  int e = blockIdx.x * blockDim.x + threadIdx.x;
  if (e >= E) return;
  int d = dst[e];
  atomicAdd(&cnt[d], 1.f);
  const float* er = ea + (size_t)e * 18;
#pragma unroll
  for (int f = 0; f < 18; ++f) atomicAdd(&ea_sum[(size_t)d * 18 + f], er[f]);
}

__global__ void loop_div(float* ea_sum, const float* cnt, int Nn)
{
  long long i = (long long)blockIdx.x * blockDim.x + threadIdx.x;
  if (i >= (long long)Nn * 18) return;
  float c = cnt[i / 18];
  ea_sum[i] = ea_sum[i] / fmaxf(c, 1.f);
}

// ---------- GATv2 attention logits (one wave per full edge), fused ea@We ----------
__global__ void gat_logits(const int* __restrict__ src, const int* __restrict__ dst,
                           const float* __restrict__ ea, const float* __restrict__ loop_ea,
                           const float* __restrict__ hl, const float* __restrict__ hr,
                           const float* __restrict__ We, const float* __restrict__ att,
                           float* logit, unsigned* menc, int E, int Nn, int F)
{
  extern __shared__ float sm[];            // [18*F] We, then [F] att
  float* sWe = sm;
  float* sAtt = sm + 18 * F;
  for (int i = threadIdx.x; i < 18 * F; i += blockDim.x) sWe[i] = We[i];
  for (int i = threadIdx.x; i < F; i += blockDim.x) sAtt[i] = att[i];
  __syncthreads();
  int wave = threadIdx.x >> 5, lane = threadIdx.x & 31;
  long long e = (long long)blockIdx.x * (blockDim.x >> 5) + wave;
  long long EF = (long long)E + Nn;
  if (e >= EF) return;
  int s, d; const float* ear;
  if (e < E) { s = src[e]; d = dst[e]; ear = ea + (size_t)e * 18; }
  else       { s = (int)(e - E); d = s; ear = loop_ea + (size_t)(e - E) * 18; }
  float er[18];
#pragma unroll
  for (int j = 0; j < 18; ++j) er[j] = ear[j];
  const float* hls = hl + (size_t)s * F;
  const float* hrd = hr + (size_t)d * F;
  float acc = 0.f;
  for (int f = lane; f < F; f += WAVE) {
    float z = hls[f] + hrd[f];
#pragma unroll
    for (int j = 0; j < 18; ++j) z += er[j] * sWe[j * F + f];
    z = (z > 0.f) ? z : 0.2f * z;            // LeakyReLU(0.2)
    acc += sAtt[f] * z;
  }
#pragma unroll
  for (int off = 16; off > 0; off >>= 1) acc += __shfl_xor(acc, off, WAVE);
  if (lane == 0) {
    logit[e] = acc;
    atomicMax(&menc[d], enc_f32(acc));
  }
}

// ---------- exp(logit - max) and denominator ----------
__global__ void gat_exp(const float* logit, const unsigned* menc, const int* dst,
                        float* evalb, float* den, int E, int Nn)
{
  long long e = (long long)blockIdx.x * blockDim.x + threadIdx.x;
  long long EF = (long long)E + Nn;
  if (e >= EF) return;
  int d = (e < E) ? dst[e] : (int)(e - E);
  float v = expf(logit[e] - dec_f32(menc[d]));
  evalb[e] = v;
  atomicAdd(&den[d], v);
}

// ---------- weighted scatter: out[d] += alpha * hl[s] ----------
__global__ void gat_scatter(const int* __restrict__ src, const int* __restrict__ dst,
                            const float* __restrict__ evalb, const float* __restrict__ den,
                            const float* __restrict__ hl, float* out, int E, int Nn, int F)
{
  int wave = threadIdx.x >> 5, lane = threadIdx.x & 31;
  long long e = (long long)blockIdx.x * (blockDim.x >> 5) + wave;
  long long EF = (long long)E + Nn;
  if (e >= EF) return;
  int s, d;
  if (e < E) { s = src[e]; d = dst[e]; } else { s = (int)(e - E); d = s; }
  float alpha = evalb[e] / den[d];
  const float* hls = hl + (size_t)s * F;
  float* od = out + (size_t)d * F;
  for (int f = lane; f < F; f += WAVE) atomicAdd(&od[f], alpha * hls[f]);
}

// ---------- Y = relu(X + bias[f]) ----------
__global__ void bias_relu(const float* X, const float* b, float* Y, long long M, int F)
{
  long long i = (long long)blockIdx.x * blockDim.x + threadIdx.x;
  if (i >= M * (long long)F) return;
  int f = (int)(i % F);
  Y[i] = fmaxf(X[i] + b[f], 0.f);
}

// ---------- BatchNorm stats (sum, sumsq per feature) ----------
__global__ void bn_stats(const float* X, float* sums, float* sqs, long long M, int F)
{
  extern __shared__ float s[];
  for (int i = threadIdx.x; i < 2 * F; i += blockDim.x) s[i] = 0.f;
  __syncthreads();
  long long total = M * F;
  for (long long idx = (long long)blockIdx.x * blockDim.x + threadIdx.x; idx < total;
       idx += (long long)gridDim.x * blockDim.x) {
    int f = (int)(idx % F);
    float v = X[idx];
    atomicAdd(&s[f], v);
    atomicAdd(&s[F + f], v * v);
  }
  __syncthreads();
  for (int f = threadIdx.x; f < F; f += blockDim.x) {
    atomicAdd(&sums[f], s[f]);
    atomicAdd(&sqs[f], s[F + f]);
  }
}

__global__ void bn_apply(const float* X, const float* sums, const float* sqs,
                         const float* g, const float* b, float* Y, long long M, int F)
{
  long long i = (long long)blockIdx.x * blockDim.x + threadIdx.x;
  if (i >= M * (long long)F) return;
  int f = (int)(i % F);
  float mu = sums[f] / (float)M;
  float var = sqs[f] / (float)M - mu * mu;
  Y[i] = (X[i] - mu) * rsqrtf(var + 1e-5f) * g[f] + b[f];
}

// ---------- global mean pool ----------
__global__ void pool_cnt(const int* batch, float* gcnt, int Nn)
{
  int i = blockIdx.x * blockDim.x + threadIdx.x;
  if (i >= Nn) return;
  atomicAdd(&gcnt[batch[i]], 1.f);
}
__global__ void pool_sum(const float* X, const int* batch, float* pool, long long Nn, int F)
{
  long long total = Nn * F;
  for (long long idx = (long long)blockIdx.x * blockDim.x + threadIdx.x; idx < total;
       idx += (long long)gridDim.x * blockDim.x) {
    int row = (int)(idx / F), f = (int)(idx % F);
    atomicAdd(&pool[(size_t)batch[row] * F + f], X[idx]);
  }
}
__global__ void pool_div(float* pool, const float* gcnt, int G, int F)
{
  long long i = (long long)blockIdx.x * blockDim.x + threadIdx.x;
  if (i >= (long long)G * F) return;
  pool[i] = pool[i] / fmaxf(gcnt[i / F], 1.f);
}

// =====================================================================================

extern "C" void kernel_launch(void* const* d_in, const int* in_sizes, int n_in,
                              void* d_out, int out_size, void* d_ws, size_t ws_size,
                              hipStream_t stream)
{
  const int N = 100000, E = 1280000, G = 1000;
  const long long EFull = (long long)E + N;

  // -------- inputs --------
  const float* x     = (const float*)d_in[0];
  const float* eattr = (const float*)d_in[1];
  const int*   eidx  = (const int*)d_in[2];
  const int*   batch = (const int*)d_in[3];
  const int* srcp = eidx;
  const int* dstp = eidx + E;
  const float* Wl1 = (const float*)d_in[4];  const float* bl1 = (const float*)d_in[5];
  const float* Wr1 = (const float*)d_in[6];  const float* br1 = (const float*)d_in[7];
  const float* We1 = (const float*)d_in[8];  const float* at1 = (const float*)d_in[9];
  const float* bo1 = (const float*)d_in[10]; const float* g1  = (const float*)d_in[11];
  const float* be1 = (const float*)d_in[12];
  const float* Wl2 = (const float*)d_in[13]; const float* bl2 = (const float*)d_in[14];
  const float* Wr2 = (const float*)d_in[15]; const float* br2 = (const float*)d_in[16];
  const float* We2 = (const float*)d_in[17]; const float* at2 = (const float*)d_in[18];
  const float* bo2 = (const float*)d_in[19]; const float* g2  = (const float*)d_in[20];
  const float* be2 = (const float*)d_in[21];
  const float* W3  = (const float*)d_in[22]; const float* b3  = (const float*)d_in[23];
  const float* g3  = (const float*)d_in[24]; const float* be3 = (const float*)d_in[25];
  const float* gp  = (const float*)d_in[26]; const float* bp  = (const float*)d_in[27];
  const float* Wf1 = (const float*)d_in[28]; const float* bf1 = (const float*)d_in[29];
  const float* Wf2 = (const float*)d_in[30]; const float* bf2 = (const float*)d_in[31];
  const float* gf  = (const float*)d_in[32]; const float* bfb = (const float*)d_in[33];
  const float* Wf3 = (const float*)d_in[34]; const float* bf3 = (const float*)d_in[35];

  // -------- workspace layout (float units, 256B-aligned regions) --------
  float* wsp = (float*)d_ws;
  auto carve = [&](long long nfloats) {
    float* p = wsp;
    wsp += (nfloats + 63) & ~63LL;   // 64 floats = 256 bytes
    return p;
  };
  float* h       = carve((long long)N * 400);   // node features (reused each stage)
  float* tmp     = carve((long long)N * 400);   // GAT accum / linear3 out
  float* hl      = carve((long long)N * 200);
  float* hr      = carve((long long)N * 200);
  float* loop_ea = carve((long long)N * 18);    // ea_sum then loop_ea in place
  float* cnt     = carve(N);
  float* logit   = carve(EFull);
  float* evalb   = carve(EFull);
  unsigned* menc = (unsigned*)carve(N);
  float* den     = carve(N);
  float* stats   = carve(800);                  // sums[0..F), sqs at +400
  float* pool    = carve((long long)G * 400);
  float* gcnt    = carve(G);
  float* f1      = carve((long long)G * 200);
  float* f2      = carve((long long)G * 100);
  __bf16* Apk    = (__bf16*)carve(12000000);    // 24M bf16 (max 6250*7 frags * 512)
  __bf16* Bpk    = (__bf16*)carve(200000);      // 400K bf16 (max 25*13 frags * 512)

  auto cdiv = [](long long a, long long b) { return (int)((a + b - 1) / b); };
  auto fillz = [&](float* p, long long n) {
    fill_f32<<<cdiv(n, 256), 256, 0, stream>>>(p, 0.f, n);
  };
  auto gemm = [&](const float* A, const float* B, const float* bias, float* C,
                  int M, int Nc, int K, int act) {
    int tilesK = (K + 31) / 32;
    int tilesM = (M + 15) / 16;
    int tilesN = (Nc + 15) / 16;
    long long ta = (long long)tilesM * tilesK * 512;
    long long tb = (long long)tilesN * tilesK * 512;
    pack_a_frag<<<cdiv(ta, 256), 256, 0, stream>>>(A, Apk, M, K, tilesK, ta);
    pack_b_frag<<<cdiv(tb, 256), 256, 0, stream>>>(B, Bpk, K, Nc, tilesK, tb);
    long long tiles = (long long)tilesM * tilesN;
    gemm_wmma_packed<<<cdiv(tiles, 4), 128, 0, stream>>>((const v16bf*)Apk,
                                                         (const v16bf*)Bpk,
                                                         bias, C, M, Nc, tilesK, act);
  };
  auto bn = [&](float* X, const float* gg, const float* bb, long long M, int F) {
    fillz(stats, 800);
    long long total = M * F;
    int blocks = cdiv(total, 256); if (blocks > 2048) blocks = 2048;
    bn_stats<<<blocks, 256, 2 * F * sizeof(float), stream>>>(X, stats, stats + 400, M, F);
    bn_apply<<<cdiv(total, 256), 256, 0, stream>>>(X, stats, stats + 400, gg, bb, X, M, F);
  };

  // -------- self-loop edge-attr means (fill_value='mean') --------
  fillz(cnt, N);
  fillz(loop_ea, (long long)N * 18);
  edge_count_sum<<<cdiv(E, 256), 256, 0, stream>>>(dstp, eattr, loop_ea, cnt, E);
  loop_div<<<cdiv((long long)N * 18, 256), 256, 0, stream>>>(loop_ea, cnt, N);

  // -------- one GATv2 layer --------
  auto gat = [&](const float* hin, int Fin, int Fout,
                 const float* Wl, const float* bl, const float* Wr, const float* br,
                 const float* Wep, const float* attp, const float* bop,
                 const float* gbn, const float* bbn) {
    gemm(hin, Wl, bl, hl, N, Fout, Fin, 0);
    gemm(hin, Wr, br, hr, N, Fout, Fin, 0);
    fillz((float*)menc, N);
    fillz(den, N);
    int lb = cdiv(EFull, 4);
    size_t shm = (size_t)(18 * Fout + Fout) * sizeof(float);
    gat_logits<<<lb, 128, shm, stream>>>(srcp, dstp, eattr, loop_ea, hl, hr, Wep, attp,
                                         logit, menc, E, N, Fout);
    gat_exp<<<cdiv(EFull, 256), 256, 0, stream>>>(logit, menc, dstp, evalb, den, E, N);
    fillz(tmp, (long long)N * Fout);
    gat_scatter<<<lb, 128, 0, stream>>>(srcp, dstp, evalb, den, hl, tmp, E, N, Fout);
    bias_relu<<<cdiv((long long)N * Fout, 256), 256, 0, stream>>>(tmp, bop, h, N, Fout);
    bn(h, gbn, bbn, N, Fout);
  };

  // -------- layer 1: 16 -> 100 --------
  gat(x, 16, 100, Wl1, bl1, Wr1, br1, We1, at1, bo1, g1, be1);
  // -------- layer 2: 100 -> 200 --------
  gat(h, 100, 200, Wl2, bl2, Wr2, br2, We2, at2, bo2, g2, be2);

  // -------- linear 200 -> 400 + BN --------
  gemm(h, W3, b3, tmp, N, 400, 200, 0);
  bn(tmp, g3, be3, N, 400);

  // -------- global mean pool + BN --------
  fillz(pool, (long long)G * 400);
  fillz(gcnt, G);
  pool_cnt<<<cdiv(N, 256), 256, 0, stream>>>(batch, gcnt, N);
  {
    long long total = (long long)N * 400;
    int blocks = cdiv(total, 256); if (blocks > 65535) blocks = 65535;
    pool_sum<<<blocks, 256, 0, stream>>>(tmp, batch, pool, N, 400);
  }
  pool_div<<<cdiv((long long)G * 400, 256), 256, 0, stream>>>(pool, gcnt, G, 400);
  bn(pool, gp, bp, G, 400);

  // -------- FFN 400 -> 200 -> 100 -> BN -> 100 --------
  gemm(pool, Wf1, bf1, f1, G, 200, 400, 1);
  gemm(f1, Wf2, bf2, f2, G, 100, 200, 1);
  bn(f2, gf, bfb, G, 100);
  gemm(f2, Wf3, bf3, (float*)d_out, G, 100, 100, 0);
}
